// PhaseTrackerLarge_36833639530538
// MI455X (gfx1250) — compile-verified, matrix operand order checked by reading, not run on hardware
//
#include <hip/hip_runtime.h>
#include <hip/hip_bf16.h>
#include <math.h>

// ---------------------------------------------------------------------------
// PhaseTracker on MI455X (gfx1250):
//   * amp branch is dead for outputs -> skipped
//   * sim GEMM (8192x224 @ 224x8192) via v_wmma_f32_16x16x32_f16,
//     32x32 register-blocked tile per wave (2x2 fragments, 4 accumulators)
//   * greedy match parallelized via packed-key atomicMax per column
// ---------------------------------------------------------------------------

#define N_DET    8192
#define D_IN     4
#define H_DIM    192
#define O_DIM    112
#define K_FEAT   224          // cos|sin concatenated
#define ROWS     16           // rows per block in the MLP kernels

#define TWO_PI_F      6.28318530717958647692f
#define INV_TWO_PI_F  0.15915494309189533577f
#define THRESH_F      0.1f

typedef _Float16 v16h __attribute__((ext_vector_type(16)));
typedef _Float16 v8h  __attribute__((ext_vector_type(8)));
typedef float    v8f  __attribute__((ext_vector_type(8)));

__device__ __forceinline__ float mod2pi(float p) {
    return p - floorf(p * INV_TWO_PI_F) * TWO_PI_F;
}

__device__ __forceinline__ v16h load_frag(const _Float16* p, int kc) {
    const v8h lo = *(const v8h*)(p + kc);
    const v8h hi = *(const v8h*)(p + kc + 16);
    return __builtin_shufflevector(lo, hi, 0,1,2,3,4,5,6,7,8,9,10,11,12,13,14,15);
}

// ---------------------------------------------------------------------------
// Encoder MLP: x(N,4) -> relu(xW1+b1) -> relu(hW2+b2) -> (hW3+b3) mod 2pi
// mode==0: write float phase (for det_t, evolve needs it)
// mode==1: write f16 [cos|sin] feature rows (for det_t1, feeds WMMA directly)
// ---------------------------------------------------------------------------
__global__ void __launch_bounds__(192)
enc_kernel(const float* __restrict__ x,
           const float* __restrict__ W1, const float* __restrict__ b1,
           const float* __restrict__ W2, const float* __restrict__ b2,
           const float* __restrict__ W3, const float* __restrict__ b3,
           float* __restrict__ out_phase, _Float16* __restrict__ out_feat,
           int write_feat)
{
    __shared__ float sx[ROWS * D_IN];
    __shared__ float h1[ROWS * H_DIM];
    __shared__ float h2[ROWS * H_DIM];

    const int j    = threadIdx.x;            // 0..191
    const int row0 = blockIdx.x * ROWS;

    for (int i = j; i < ROWS * D_IN; i += H_DIM)
        sx[i] = x[row0 * D_IN + i];
    __syncthreads();

    // ---- layer 1: (4 -> 192) + ReLU ----
    {
        const float w0 = W1[0 * H_DIM + j], w1 = W1[1 * H_DIM + j];
        const float w2 = W1[2 * H_DIM + j], w3 = W1[3 * H_DIM + j];
        const float bb = b1[j];
        #pragma unroll
        for (int r = 0; r < ROWS; ++r) {
            float v = bb;
            v = fmaf(sx[r * 4 + 0], w0, v);
            v = fmaf(sx[r * 4 + 1], w1, v);
            v = fmaf(sx[r * 4 + 2], w2, v);
            v = fmaf(sx[r * 4 + 3], w3, v);
            h1[r * H_DIM + j] = v > 0.0f ? v : 0.0f;
        }
    }
    __syncthreads();

    // ---- layer 2: (192 -> 192) + ReLU, float4 LDS reads ----
    {
        float acc[ROWS];
        const float bb = b2[j];
        #pragma unroll
        for (int r = 0; r < ROWS; ++r) acc[r] = bb;
        for (int k = 0; k < H_DIM; k += 4) {
            const float w0 = W2[(k + 0) * H_DIM + j];
            const float w1 = W2[(k + 1) * H_DIM + j];
            const float w2 = W2[(k + 2) * H_DIM + j];
            const float w3 = W2[(k + 3) * H_DIM + j];
            #pragma unroll
            for (int r = 0; r < ROWS; ++r) {
                const float4 hv = *reinterpret_cast<const float4*>(&h1[r * H_DIM + k]);
                acc[r] = fmaf(hv.x, w0, acc[r]);
                acc[r] = fmaf(hv.y, w1, acc[r]);
                acc[r] = fmaf(hv.z, w2, acc[r]);
                acc[r] = fmaf(hv.w, w3, acc[r]);
            }
        }
        #pragma unroll
        for (int r = 0; r < ROWS; ++r)
            h2[r * H_DIM + j] = acc[r] > 0.0f ? acc[r] : 0.0f;
    }
    __syncthreads();

    // ---- layer 3: (192 -> 112), mod 2pi, emit phase or features ----
    if (j < O_DIM) {
        float acc[ROWS];
        const float bb = b3[j];
        #pragma unroll
        for (int r = 0; r < ROWS; ++r) acc[r] = bb;
        for (int k = 0; k < H_DIM; k += 4) {
            const float w0 = W3[(k + 0) * O_DIM + j];
            const float w1 = W3[(k + 1) * O_DIM + j];
            const float w2 = W3[(k + 2) * O_DIM + j];
            const float w3 = W3[(k + 3) * O_DIM + j];
            #pragma unroll
            for (int r = 0; r < ROWS; ++r) {
                const float4 hv = *reinterpret_cast<const float4*>(&h2[r * H_DIM + k]);
                acc[r] = fmaf(hv.x, w0, acc[r]);
                acc[r] = fmaf(hv.y, w1, acc[r]);
                acc[r] = fmaf(hv.z, w2, acc[r]);
                acc[r] = fmaf(hv.w, w3, acc[r]);
            }
        }
        #pragma unroll
        for (int r = 0; r < ROWS; ++r) {
            const float p = mod2pi(acc[r]);
            if (write_feat) {
                const size_t base = (size_t)(row0 + r) * K_FEAT;
                out_feat[base + j]         = (_Float16)cosf(p);
                out_feat[base + O_DIM + j] = (_Float16)sinf(p);
            } else {
                out_phase[(size_t)(row0 + r) * O_DIM + j] = p;
            }
        }
    }
}

// ---------------------------------------------------------------------------
// Evolve: p += omega*0.05 ; p += 0.1*(relu(p@Wr1+br1)@Wr2+br2) ; mod 2pi
// Emits f16 [cos|sin] features for the evolved phases.
// ---------------------------------------------------------------------------
__global__ void __launch_bounds__(192)
evolve_kernel(const float* __restrict__ phase_t, const float* __restrict__ omega,
              const float* __restrict__ Wr1, const float* __restrict__ br1,
              const float* __restrict__ Wr2, const float* __restrict__ br2,
              _Float16* __restrict__ out_feat)
{
    __shared__ float sp[ROWS * O_DIM];
    __shared__ float ract[ROWS * H_DIM];

    const int j    = threadIdx.x;
    const int row0 = blockIdx.x * ROWS;

    if (j < O_DIM) {
        const float om = omega[j] * 0.05f;   // N_STEPS * DT
        #pragma unroll
        for (int r = 0; r < ROWS; ++r)
            sp[r * O_DIM + j] = phase_t[(size_t)(row0 + r) * O_DIM + j] + om;
    }
    __syncthreads();

    // relu(p @ Wr1 + br1): (112 -> 192)
    {
        float acc[ROWS];
        const float bb = br1[j];
        #pragma unroll
        for (int r = 0; r < ROWS; ++r) acc[r] = bb;
        for (int k = 0; k < O_DIM; k += 4) {
            const float w0 = Wr1[(k + 0) * H_DIM + j];
            const float w1 = Wr1[(k + 1) * H_DIM + j];
            const float w2 = Wr1[(k + 2) * H_DIM + j];
            const float w3 = Wr1[(k + 3) * H_DIM + j];
            #pragma unroll
            for (int r = 0; r < ROWS; ++r) {
                const float4 hv = *reinterpret_cast<const float4*>(&sp[r * O_DIM + k]);
                acc[r] = fmaf(hv.x, w0, acc[r]);
                acc[r] = fmaf(hv.y, w1, acc[r]);
                acc[r] = fmaf(hv.z, w2, acc[r]);
                acc[r] = fmaf(hv.w, w3, acc[r]);
            }
        }
        #pragma unroll
        for (int r = 0; r < ROWS; ++r)
            ract[r * H_DIM + j] = acc[r] > 0.0f ? acc[r] : 0.0f;
    }
    __syncthreads();

    // p + 0.1*(ract @ Wr2 + br2), mod 2pi, feature emit: (192 -> 112)
    if (j < O_DIM) {
        float acc[ROWS];
        const float bb = br2[j];
        #pragma unroll
        for (int r = 0; r < ROWS; ++r) acc[r] = bb;
        for (int k = 0; k < H_DIM; k += 4) {
            const float w0 = Wr2[(k + 0) * O_DIM + j];
            const float w1 = Wr2[(k + 1) * O_DIM + j];
            const float w2 = Wr2[(k + 2) * O_DIM + j];
            const float w3 = Wr2[(k + 3) * O_DIM + j];
            #pragma unroll
            for (int r = 0; r < ROWS; ++r) {
                const float4 hv = *reinterpret_cast<const float4*>(&ract[r * H_DIM + k]);
                acc[r] = fmaf(hv.x, w0, acc[r]);
                acc[r] = fmaf(hv.y, w1, acc[r]);
                acc[r] = fmaf(hv.z, w2, acc[r]);
                acc[r] = fmaf(hv.w, w3, acc[r]);
            }
        }
        #pragma unroll
        for (int r = 0; r < ROWS; ++r) {
            const float p = mod2pi(sp[r * O_DIM + j] + 0.1f * acc[r]);
            const size_t base = (size_t)(row0 + r) * K_FEAT;
            out_feat[base + j]         = (_Float16)cosf(p);
            out_feat[base + O_DIM + j] = (_Float16)sinf(p);
        }
    }
}

// ---------------------------------------------------------------------------
// Similarity GEMM via WMMA: sim = featA(8192x224,f16) @ featB^T, scaled.
// 32x32 output tile per wave: 2 A frags x 2 B frags x 4 f32 accumulators.
// K=224 -> 7 k-chunks; per chunk 8 x b128 loads feed 4 WMMAs (2:1).
// Fragment layout per the ISA 16-bit 16x32 A-matrix table:
//   lane<16 holds row (lane&15), K {kc..kc+7, kc+16..kc+23};
//   lane>=16 same row, K {kc+8..kc+15, kc+24..kc+31}. B is symmetric.
// ---------------------------------------------------------------------------
__global__ void __launch_bounds__(256)
sim_wmma_kernel(const _Float16* __restrict__ featA,
                const _Float16* __restrict__ featB,
                float* __restrict__ sim)
{
    const int lane = threadIdx.x & 31;
    const int wave = threadIdx.x >> 5;
    const unsigned tile = blockIdx.x * 8u + wave;   // 256x256 32x32-tiles
    const int tm   = tile >> 8;
    const int tn   = tile & 255;
    const int half = lane >> 4;     // K-subgroup select
    const int sub  = lane & 15;

    const _Float16* ap0 = featA + (size_t)(tm * 32 + sub) * K_FEAT + half * 8;
    const _Float16* ap1 = ap0 + (size_t)16 * K_FEAT;
    const _Float16* bp0 = featB + (size_t)(tn * 32 + sub) * K_FEAT + half * 8;
    const _Float16* bp1 = bp0 + (size_t)16 * K_FEAT;

    v8f acc00 = {}, acc01 = {}, acc10 = {}, acc11 = {};
    #pragma unroll
    for (int kc = 0; kc < K_FEAT; kc += 32) {
        const v16h a0 = load_frag(ap0, kc);
        const v16h a1 = load_frag(ap1, kc);
        const v16h b0 = load_frag(bp0, kc);
        const v16h b1 = load_frag(bp1, kc);
        acc00 = __builtin_amdgcn_wmma_f32_16x16x32_f16(false, a0, false, b0,
                                                       (short)0, acc00, false, false);
        acc01 = __builtin_amdgcn_wmma_f32_16x16x32_f16(false, a0, false, b1,
                                                       (short)0, acc01, false, false);
        acc10 = __builtin_amdgcn_wmma_f32_16x16x32_f16(false, a1, false, b0,
                                                       (short)0, acc10, false, false);
        acc11 = __builtin_amdgcn_wmma_f32_16x16x32_f16(false, a1, false, b1,
                                                       (short)0, acc11, false, false);
    }

    const float sq    = sqrtf((float)O_DIM) + 1e-8f;
    const float scale = 1.0f / (sq * sq);

    // C layout: VGPR v, lanes 0-15 -> M=v ; lanes 16-31 -> M=v+8 ; N=sub
    const size_t rm = (size_t)tm * 32 + half * 8;
    const size_t cn = (size_t)tn * 32 + sub;
    float* out00 = sim + (rm +  0) * N_DET + cn;        // A0 x B0
    float* out01 = out00 + 16;                          // A0 x B1
    float* out10 = sim + (rm + 16) * N_DET + cn;        // A1 x B0
    float* out11 = out10 + 16;                          // A1 x B1
    #pragma unroll
    for (int v = 0; v < 8; ++v) {
        out00[(size_t)v * N_DET] = acc00[v] * scale;
        out01[(size_t)v * N_DET] = acc01[v] * scale;
        out10[(size_t)v * N_DET] = acc10[v] * scale;
        out11[(size_t)v * N_DET] = acc11[v] * scale;
    }
}

// ---------------------------------------------------------------------------
// Greedy match, parallelized. winner[j] holds packed (orderable-sim-key<<32 |
// (0xFFFFFFFF - row)) of the best row targeting column j (the row the
// sequential greedy would have given j to).
// ---------------------------------------------------------------------------
__global__ void __launch_bounds__(256)
init_winner_kernel(unsigned long long* __restrict__ winner)
{
    const int i = blockIdx.x * blockDim.x + threadIdx.x;
    if (i < N_DET) winner[i] = 0ull;
}

__global__ void __launch_bounds__(256)
rowmax_claim_kernel(const float* __restrict__ sim,
                    float* __restrict__ max_sims, int* __restrict__ max_idxs,
                    unsigned long long* __restrict__ winner)
{
    __shared__ float sm[256];
    __shared__ int   si[256];
    const int row = blockIdx.x;
    const int t   = threadIdx.x;
    const float* srow = sim + (size_t)row * N_DET;

    float m  = -3.0e38f;
    int   mi = 0;
    for (int c = t; c < N_DET; c += 256) {
        const float v = srow[c];
        if (v > m) { m = v; mi = c; }
    }
    sm[t] = m; si[t] = mi;
    __syncthreads();
    for (int s = 128; s > 0; s >>= 1) {
        if (t < s) {
            const float om = sm[t + s]; const int oi = si[t + s];
            if (om > sm[t] || (om == sm[t] && oi < si[t])) { sm[t] = om; si[t] = oi; }
        }
        __syncthreads();
    }
    if (t == 0) {
        m = sm[0]; mi = si[0];
        max_sims[row] = m;
        max_idxs[row] = mi;
        if (m >= THRESH_F) {
            unsigned k = __float_as_uint(m);
            k = (k & 0x80000000u) ? ~k : (k | 0x80000000u);   // total-order key
            const unsigned long long pack =
                ((unsigned long long)k << 32) | (unsigned long long)(0xFFFFFFFFu - (unsigned)row);
            atomicMax(&winner[mi], pack);
        }
    }
}

__global__ void __launch_bounds__(256)
resolve_kernel(const float* __restrict__ max_sims, const int* __restrict__ max_idxs,
               const unsigned long long* __restrict__ winner, int* __restrict__ matches)
{
    const int r = blockIdx.x * blockDim.x + threadIdx.x;
    if (r >= N_DET) return;
    int match = -1;
    const float m = max_sims[r];
    if (m >= THRESH_F) {
        const int j = max_idxs[r];
        if ((unsigned)(winner[j] & 0xFFFFFFFFull) == 0xFFFFFFFFu - (unsigned)r)
            match = j;
    }
    matches[r] = match;
}

// ---------------------------------------------------------------------------
// Workspace layout (bytes)
// ---------------------------------------------------------------------------
#define WS_FEATA   ((size_t)0)                         // 8192*224*2 = 3670016
#define WS_FEATB   ((size_t)3670016)
#define WS_PHASET  ((size_t)7340032)                   // 8192*112*4 = 3670016
#define WS_MAXS    ((size_t)11010048)                  // 8192*4
#define WS_MAXI    (WS_MAXS + 32768)                   // 8192*4
#define WS_WINNER  (WS_MAXS + 65536)                   // 8192*8

extern "C" void kernel_launch(void* const* d_in, const int* in_sizes, int n_in,
                              void* d_out, int out_size, void* d_ws, size_t ws_size,
                              hipStream_t stream) {
    const float* det_t  = (const float*)d_in[0];
    const float* det_t1 = (const float*)d_in[1];
    const float* W1 = (const float*)d_in[2];   const float* b1 = (const float*)d_in[3];
    const float* W2 = (const float*)d_in[4];   const float* b2 = (const float*)d_in[5];
    const float* W3 = (const float*)d_in[6];   const float* b3 = (const float*)d_in[7];
    // d_in[8..11] = Wa1/ba1/Wa2/ba2: dead for outputs (amp unused by matches/sim)
    const float* Wr1 = (const float*)d_in[12]; const float* br1 = (const float*)d_in[13];
    const float* Wr2 = (const float*)d_in[14]; const float* br2 = (const float*)d_in[15];
    const float* omega = (const float*)d_in[16];

    char* ws = (char*)d_ws;
    _Float16* featA = (_Float16*)(ws + WS_FEATA);
    _Float16* featB = (_Float16*)(ws + WS_FEATB);
    float*    phase_t  = (float*)(ws + WS_PHASET);
    float*    max_sims = (float*)(ws + WS_MAXS);
    int*      max_idxs = (int*)(ws + WS_MAXI);
    unsigned long long* winner = (unsigned long long*)(ws + WS_WINNER);

    int*   matches = (int*)d_out;               // output 0: 8192 x int32
    float* sim     = (float*)d_out + N_DET;     // output 1: 8192x8192 f32

    const int mlp_blocks = N_DET / ROWS;        // 512

    // encode det_t -> float phase_t
    enc_kernel<<<mlp_blocks, 192, 0, stream>>>(det_t, W1, b1, W2, b2, W3, b3,
                                               phase_t, featA, 0);
    // encode det_t1 -> f16 features B
    enc_kernel<<<mlp_blocks, 192, 0, stream>>>(det_t1, W1, b1, W2, b2, W3, b3,
                                               phase_t /*unused*/, featB, 1);
    // evolve phase_t -> f16 features A
    evolve_kernel<<<mlp_blocks, 192, 0, stream>>>(phase_t, omega, Wr1, br1, Wr2, br2,
                                                  featA);
    // greedy-match winner table init
    init_winner_kernel<<<N_DET / 256, 256, 0, stream>>>(winner);
    // sim GEMM: 256x256 32x32-tiles, 8 waves/block
    sim_wmma_kernel<<<(256 * 256) / 8, 256, 0, stream>>>(featA, featB, sim);
    // per-row max/argmax + atomic claim
    rowmax_claim_kernel<<<N_DET, 256, 0, stream>>>(sim, max_sims, max_idxs, winner);
    // resolve matches
    resolve_kernel<<<N_DET / 256, 256, 0, stream>>>(max_sims, max_idxs, winner, matches);
}
